// RotConv_41712722379541
// MI455X (gfx1250) — matrix operand order repaired and checked by reading, not compiled, not run on hardware
//
#include <hip/hip_runtime.h>
#include <hip/hip_bf16.h>

typedef __attribute__((ext_vector_type(16))) _Float16 v16h;
typedef __attribute__((ext_vector_type(8)))  float    v8f;

#define R_ROT   32          // 8 inclinations x 4 azimuths
#define OCH     16
#define ICH     8
#define KPAD    224         // 216 (=8*27) padded to 7 * 32
#define KSTEPS  7
#define NPOS    64000       // 40*40*40
#define WAVES_PER_BLOCK 4   // 128 threads; each wave owns 32 voxels (2 M-tiles)

// ---------------------------------------------------------------------------
// Kernel 1: build rotated weights directly in WMMA B-fragment layout (f16).
// Wswz[((r*7+s)*32 + lane)*16 + h] = W_logical[K = 32*s + (lane>=16?16:0) + h]
//                                             [N = r*16 + (lane&15)]
// W_logical[(i,p)][(r,o)] = sum_q T_rot[r][p][q] * weight1[o][i][q]
// T_rot row (r,p) has <= 8 nonzeros (trilinear corners) -> computed inline.
// ---------------------------------------------------------------------------
__global__ void build_wswz(const float* __restrict__ weight1,
                           _Float16* __restrict__ Wswz) {
    int gid = blockIdx.x * blockDim.x + threadIdx.x;   // 0 .. 114687
    int h    = gid & 15;
    int t    = gid >> 4;
    int lane = t & 31;
    int u    = t >> 5;          // 0..223
    int s    = u % KSTEPS;
    int r    = u / KSTEPS;      // 0..31

    int o = lane & 15;
    int k = 32 * s + ((lane >= 16) ? 16 : 0) + h;   // logical K index

    float val = 0.0f;
    if (k < ICH * 27) {
        int i = k / 27;
        int p = k % 27;
        // rotation r = ti*4 + pj ; theta = ti * pi/8 ; phi = pj * pi/2
        int ti = r >> 2, pj = r & 3;
        float theta = (float)ti * 0.39269908169872414f;  // pi/8
        float phi   = (float)pj * 1.5707963267948966f;   // pi/2
        float ct = cosf(theta), st = sinf(theta);
        float cp = cosf(phi),   sp = sinf(phi);
        // R = Rz @ Ry
        float R00 = cp * ct, R01 = -sp, R02 = cp * st;
        float R10 = sp * ct, R11 =  cp, R12 = sp * st;
        float R20 = -st,     R21 = 0.f, R22 = ct;
        // dst coords (z,y,x) centered at 1
        float d0 = (float)(p / 9)       - 1.0f;
        float d1 = (float)((p / 3) % 3) - 1.0f;
        float d2 = (float)(p % 3)       - 1.0f;
        // src = (d - c) @ R + c   (row-vector times R)
        float s0 = d0 * R00 + d1 * R10 + d2 * R20 + 1.0f;
        float s1 = d0 * R01 + d1 * R11 + d2 * R21 + 1.0f;
        float s2 = d0 * R02 + d1 * R12 + d2 * R22 + 1.0f;
        float f0 = floorf(s0), f1 = floorf(s1), f2 = floorf(s2);
        float w0 = s0 - f0, w1 = s1 - f1, w2 = s2 - f2;
        int i0 = (int)f0, i1 = (int)f1, i2 = (int)f2;
        const float* wbase = weight1 + (o * ICH + i) * 27;
        #pragma unroll
        for (int dz = 0; dz < 2; ++dz)
        #pragma unroll
        for (int dy = 0; dy < 2; ++dy)
        #pragma unroll
        for (int dx = 0; dx < 2; ++dx) {
            int n0 = i0 + dz, n1 = i1 + dy, n2 = i2 + dx;
            float wt = (dz ? w0 : 1.0f - w0)
                     * (dy ? w1 : 1.0f - w1)
                     * (dx ? w2 : 1.0f - w2);
            if (n0 >= 0 && n0 < 3 && n1 >= 0 && n1 < 3 && n2 >= 0 && n2 < 3) {
                int q = n0 * 9 + n1 * 3 + n2;
                val += wt * wbase[q];
            }
        }
    }
    Wswz[gid] = (_Float16)val;
}

// ---------------------------------------------------------------------------
// Kernel 2: fused im2col + WMMA GEMM + max/argmax over rotations + trig.
// One wave handles 32 consecutive voxels (two 16-row M-tiles) x 512 filters:
// each B fragment is loaded once and consumed by two independent WMMA chains.
// ---------------------------------------------------------------------------
__global__ __launch_bounds__(WAVES_PER_BLOCK * 32)
void rotconv(const float* __restrict__ x,
             const _Float16* __restrict__ Wswz,
             float* __restrict__ out) {
    // Per-wave A staging: 2 M-tiles * 7 ksteps * 32 lanes * v16h = 14 KB/wave
    __shared__ v16h lds[WAVES_PER_BLOCK * 2 * KSTEPS * 32];

    int lane = threadIdx.x & 31;
    int wid  = threadIdx.x >> 5;
    int tile = blockIdx.x * WAVES_PER_BLOCK + wid;    // 0..3999 exactly
    int b    = tile / 2000;
    int spatialBase = (tile - b * 2000) * 32;         // voxel index within batch

    const float* xb = x + (long)b * ICH * NPOS;
    _Float16* aw = (_Float16*)(lds + wid * 2 * KSTEPS * 32);

    // ---- im2col gather into LDS, written directly in A-fragment layout ----
    // With k = q*32 + lane (q = kstep), the element->fragment-slot mapping and
    // the k decomposition are loop-invariant per lane/q; hoist them all.
    // A 16x32 f16 layout (contiguous, matches B): lane_slot = m + (hi?16:0); half h:
    //   hi=(kk>=16), h=kk-(hi?16:0)               (h in 0..15)
    int kk = lane;
    int hi, h;
    hi = (kk >= 16) ? 1 : 0; h = kk - (hi ? 16 : 0);
    int flOff = hi ? 16 : 0;

    int  qbase[KSTEPS];   // i*64000 (channel offset)
    int  qpz[KSTEPS], qpy[KSTEPS], qpx[KSTEPS];
    bool qv[KSTEPS];
    #pragma unroll
    for (int q = 0; q < KSTEPS; ++q) {
        int k = q * 32 + lane;
        if (k < ICH * 27) {
            int i = k / 27, p = k % 27;
            qv[q]    = true;
            qbase[q] = i * NPOS;
            qpz[q]   = p / 9 - 1;
            qpy[q]   = (p / 3) % 3 - 1;
            qpx[q]   = p % 3 - 1;
        } else {
            qv[q] = false; qbase[q] = 0; qpz[q] = 0; qpy[q] = 0; qpx[q] = 0;
        }
    }

    for (int m = 0; m < 32; ++m) {
        int pos = spatialBase + m;
        int d   = pos / 1600;
        int rem = pos - d * 1600;
        int hh  = rem / 40;
        int ww  = rem - hh * 40;
        int mt  = m >> 4;          // which M-tile
        int ml  = m & 15;          // row within tile
        #pragma unroll
        for (int q = 0; q < KSTEPS; ++q) {
            float val = 0.0f;
            int dd = d  + qpz[q];
            int hy = hh + qpy[q];
            int wx = ww + qpx[q];
            if (qv[q] && (unsigned)dd < 40u && (unsigned)hy < 40u &&
                (unsigned)wx < 40u)
                val = xb[qbase[q] + dd * 1600 + hy * 40 + wx];
            aw[((mt * KSTEPS + q) * 32 + ml + flOff) * 16 + h] = (_Float16)val;
        }
    }
    __syncthreads();

    // ---- load the 14 A fragments once into registers ----
    const v16h* afrag = lds + wid * 2 * KSTEPS * 32;
    v16h a0[KSTEPS], a1[KSTEPS];
    #pragma unroll
    for (int s = 0; s < KSTEPS; ++s) {
        a0[s] = afrag[s * 32 + lane];
        a1[s] = afrag[(KSTEPS + s) * 32 + lane];
    }

    // ---- rotation loop: 7 B loads -> barrier -> 14 WMMAs (2 indep chains) ----
    float best0[8], best1[8];
    int   bidx0[8], bidx1[8];
    #pragma unroll
    for (int j = 0; j < 8; ++j) {
        best0[j] = -3.402823466e38f; bidx0[j] = 0;
        best1[j] = -3.402823466e38f; bidx1[j] = 0;
    }

    const v16h* Bf = (const v16h*)Wswz;
    for (int r = 0; r < R_ROT; ++r) {
        v16h bf[KSTEPS];
        #pragma unroll
        for (int s = 0; s < KSTEPS; ++s)
            bf[s] = Bf[(r * KSTEPS + s) * 32 + lane];
        // Keep the load group together: nothing may cross this point, so all
        // 14 b128 loads issue as a clause and WMMA waits stagger down.
        __builtin_amdgcn_sched_barrier(0);

        v8f acc0 = {0.f, 0.f, 0.f, 0.f, 0.f, 0.f, 0.f, 0.f};
        v8f acc1 = {0.f, 0.f, 0.f, 0.f, 0.f, 0.f, 0.f, 0.f};
        #pragma unroll
        for (int s = 0; s < KSTEPS; ++s) {
            acc0 = __builtin_amdgcn_wmma_f32_16x16x32_f16(
                false, a0[s], false, bf[s], (short)0, acc0, false, false);
            acc1 = __builtin_amdgcn_wmma_f32_16x16x32_f16(
                false, a1[s], false, bf[s], (short)0, acc1, false, false);
        }
        #pragma unroll
        for (int j = 0; j < 8; ++j) {
            if (acc0[j] > best0[j]) { best0[j] = acc0[j]; bidx0[j] = r; }
            if (acc1[j] > best1[j]) { best1[j] = acc1[j]; bidx1[j] = r; }
        }
    }

    // ---- epilogue: C layout gives lane -> (o = lane&15, m = (lane>>4)*8+j) ----
    int o = lane & 15;
    int mbase = (lane >> 4) * 8;
    float* ub = out;
    float* vb = out + 2 * OCH * NPOS;       // 2,048,000
    float* wb = out + 4 * OCH * NPOS;       // 4,096,000
    long cbase = ((long)b * OCH + o) * NPOS + spatialBase;
    #pragma unroll
    for (int j = 0; j < 8; ++j) {
        {
            float sN  = best0[j] > 0.0f ? best0[j] : 0.0f;
            float ang = (float)bidx0[j] * 0.7853981633974483f;  // idx * pi/4
            float sa  = sinf(ang), ca = cosf(ang);
            long idx = cbase + mbase + j;
            ub[idx] = sN * sa * ca;
            vb[idx] = sN * sa * sa;
            wb[idx] = sN * ca;
        }
        {
            float sN  = best1[j] > 0.0f ? best1[j] : 0.0f;
            float ang = (float)bidx1[j] * 0.7853981633974483f;
            float sa  = sinf(ang), ca = cosf(ang);
            long idx = cbase + 16 + mbase + j;
            ub[idx] = sN * sa * ca;
            vb[idx] = sN * sa * sa;
            wb[idx] = sN * ca;
        }
    }
}

// ---------------------------------------------------------------------------
extern "C" void kernel_launch(void* const* d_in, const int* in_sizes, int n_in,
                              void* d_out, int out_size, void* d_ws, size_t ws_size,
                              hipStream_t stream) {
    const float* x  = (const float*)d_in[0];   // [2, 8, 40, 40, 40]
    const float* w1 = (const float*)d_in[1];   // [16, 8, 3, 3, 3]
    _Float16* Wswz  = (_Float16*)d_ws;         // 114688 halves = 229 KB
    float* out      = (float*)d_out;           // u|v|w, 6,144,000 floats

    // 32 rot * 7 ksteps * 32 lanes * 16 halves = 114688 entries
    build_wswz<<<114688 / 256, 256, 0, stream>>>(w1, Wswz);

    // 4000 wave-jobs (2 batches * 2000 32-voxel tiles), 4 waves per block
    rotconv<<<4000 / WAVES_PER_BLOCK, WAVES_PER_BLOCK * 32, 0, stream>>>(x, Wswz, out);
}
